// ContrastLoss_32701880992455
// MI455X (gfx1250) — matrix operand order repaired and checked by reading, compiled, not run on hardware
//
#include <hip/hip_runtime.h>
#include <hip/hip_bf16.h>

// ---- problem constants (from reference setup_inputs) ----
#define B_ 8
#define N_ 4096
#define C_ 256
#define M_ (N_ / 3)          // 1365
#define MT_ 86               // ceil(M/16)
#define MTP_ (MT_ * 16)      // 1376 padded anchor rows
#define RADIUS_ (M_ / 2)     // 682
#define NBR_ 5
#define POS_MARGIN 0.1f
#define NEG_MARGIN 1.0f

typedef __attribute__((ext_vector_type(16))) _Float16 v16h;
typedef __attribute__((ext_vector_type(8)))  _Float16 v8h;
typedef __attribute__((ext_vector_type(8)))  float    v8f;

// ---------------- reduction helpers (blockDim.x == 256, wave32) -------------
__device__ __forceinline__ float waveSum(float v) {
#pragma unroll
  for (int off = 16; off; off >>= 1) v += __shfl_xor(v, off, 32);
  return v;
}
__device__ __forceinline__ float waveMax(float v) {
#pragma unroll
  for (int off = 16; off; off >>= 1) v = fmaxf(v, __shfl_xor(v, off, 32));
  return v;
}
__device__ __forceinline__ float blockSum(float v, float* lds) {
  v = waveSum(v);
  int lane = threadIdx.x & 31, w = threadIdx.x >> 5;
  __syncthreads();
  if (lane == 0) lds[w] = v;
  __syncthreads();
  float r = lds[0];
#pragma unroll
  for (int i = 1; i < 8; ++i) r += lds[i];
  return r;
}
__device__ __forceinline__ float blockMax(float v, float* lds) {
  v = waveMax(v);
  int lane = threadIdx.x & 31, w = threadIdx.x >> 5;
  __syncthreads();
  if (lane == 0) lds[w] = v;
  __syncthreads();
  float r = lds[0];
#pragma unroll
  for (int i = 1; i < 8; ++i) r = fmaxf(r, lds[i]);
  return r;
}

// ------------------- kernel 1: y -> f16, y2 = ||y_row||^2 -------------------
__global__ void __launch_bounds__(256) prep_y(const float* __restrict__ y,
                                              _Float16* __restrict__ y16,
                                              float* __restrict__ y2) {
  __shared__ float lds[8];
  const int row = blockIdx.x;                 // b*N + n
  const int c = threadIdx.x;
  const size_t idx = (size_t)row * C_ + c;
  const float v = y[idx];
  y16[idx] = (_Float16)v;
  float s = blockSum(v * v, lds);
  if (threadIdx.x == 0) y2[row] = s;
}

// ------- kernel 2: anchor slice -> f16 (zero padded), a2, fp32 d_pos --------
__global__ void __launch_bounds__(256) prep_x(const float* __restrict__ x,
                                              const float* __restrict__ y,
                                              const int* __restrict__ stp,
                                              _Float16* __restrict__ x16,
                                              float* __restrict__ a2,
                                              float* __restrict__ dpos) {
  __shared__ float lds[8];
  const int bid = blockIdx.x;
  const int b = bid / MTP_, m = bid % MTP_;
  const int c = threadIdx.x;
  const int st = *stp;
  float v = 0.f, w = 0.f;
  if (m < M_) {
    const size_t src = ((size_t)b * N_ + (size_t)(st + m)) * C_ + c;
    v = x[src];
    w = y[src];
  }
  x16[((size_t)b * MTP_ + m) * C_ + c] = (_Float16)v;
  float s1 = blockSum(v * v, lds);
  float dd = v - w;
  float s2 = blockSum(dd * dd, lds);
  if (threadIdx.x == 0) {
    a2[b * MTP_ + m] = s1;
    if (m < M_) dpos[b * M_ + m] = sqrtf(s2);
  }
}

// ------------- kernel 3: fused score_x + score_y over used slice ------------
__device__ __forceinline__ float row_score(const float* __restrict__ base,
                                           int i, int c, float* lds) {
  const float v = base[(size_t)i * C_ + c];
  int j0 = i - NBR_; if (j0 < 0) j0 = 0;
  int j1 = i + NBR_ + 1; if (j1 > N_) j1 = N_;
  float s = 0.f;
  for (int j = j0; j < j1; ++j) s += base[(size_t)j * C_ + c];
  s -= v;                                     // exclude self
  const float avg = s / (float)(j1 - j0 - 1);
  const float mx = blockMax(v, lds);          // row max over channels
  const float t = v - avg;
  const float alpha = (t > 20.f) ? t : log1pf(expf(t));   // softplus
  const float beta = v / mx;
  return blockMax(alpha * beta, lds);
}

__global__ void __launch_bounds__(256) score_k(const float* __restrict__ x,
                                               const float* __restrict__ y,
                                               const int* __restrict__ stp,
                                               float* __restrict__ sc) {
  __shared__ float lds[8];
  const int bid = blockIdx.x;
  const int b = bid / M_, m = bid % M_;
  const int i = *stp + m;
  const int c = threadIdx.x;
  const float sx = row_score(x + (size_t)b * N_ * C_, i, c, lds);
  const float sy = row_score(y + (size_t)b * N_ * C_, i, c, lds);
  if (threadIdx.x == 0) sc[b * M_ + m] = sx + sy;
}

// ------ kernel 4: WMMA GEMM -> masked min distance (d_neg) per anchor -------
__global__ void __launch_bounds__(256) gemm_dneg(const _Float16* __restrict__ x16,
                                                 const _Float16* __restrict__ y16,
                                                 const float* __restrict__ a2,
                                                 const float* __restrict__ y2,
                                                 const int* __restrict__ stp,
                                                 float* __restrict__ dneg) {
  __shared__ float sMin[8][16];
  const int bid = blockIdx.x;
  const int b = bid / MT_, mt = bid % MT_;
  const int st = *stp;
  const int tid = threadIdx.x;
  const int wv = tid >> 5;
  const int lane = tid & 31;
  const int half = lane >> 4;                 // lane 0-15 vs 16-31
  const int lrow = lane & 15;

  // ---- preload A fragments for all 8 K-steps (row = anchor m, 16x32 layout)
  // lane half h: 16 contiguous K values (K = 16h..16h+15)
  const _Float16* xrow = x16 + ((size_t)b * MTP_ + (size_t)(mt * 16 + lrow)) * C_;
  v16h afr[8];
#pragma unroll
  for (int kk = 0; kk < 8; ++kk) {
    union { v16h v; v8h p[2]; } u;
    u.p[0] = *(const v8h*)(xrow + kk * 32 + 16 * half);
    u.p[1] = *(const v8h*)(xrow + kk * 32 + 16 * half + 8);
    afr[kk] = u.v;
  }
  // a2 for the 8 m-rows this lane's accumulator slots represent (M = 8h + r)
  float a2r[8];
  {
    const float* ap = a2 + b * MTP_ + mt * 16 + 8 * half;
#pragma unroll
    for (int r = 0; r < 8; ++r) a2r[r] = ap[r];
  }
  const _Float16* ybase = y16 + (size_t)b * N_ * C_;
  const float* y2b = y2 + b * N_;

  float mn[8];
#pragma unroll
  for (int r = 0; r < 8; ++r) mn[r] = __builtin_inff();

  // waves scan interleaved 16-column n-tiles of the full N
  for (int t = wv; t < N_ / 16; t += 8) {
    const int n0 = t * 16;
    const _Float16* yrow = ybase + (size_t)(n0 + lrow) * C_;  // column n = n0+lrow
    v8f acc = {};
#pragma unroll
    for (int kk = 0; kk < 8; ++kk) {
      // B 32x16 layout: lane holds 16 contiguous K values (K = 16h..16h+15)
      v16h bfr = *(const v16h*)(yrow + kk * 32 + 16 * half);
      acc = __builtin_amdgcn_wmma_f32_16x16x32_f16(
          false, afr[kk], false, bfr, (short)0, acc, false, false);
    }
    const int n = n0 + lrow;
    const float y2v = y2b[n];
#pragma unroll
    for (int r = 0; r < 8; ++r) {
      const int m = mt * 16 + 8 * half + r;   // C/D layout: VGPR r, lane half
      const int i = st + m;
      const float sq = a2r[r] + y2v - 2.0f * acc[r];
      const float d = sqrtf(fmaxf(sq, 0.f));
      const bool is_neg = (n < i - RADIUS_) || (n >= i + RADIUS_);
      if (is_neg) mn[r] = fminf(mn[r], d);
    }
  }
  // min across the 16 lanes sharing each m (xor of bits 0..3 stays in-half)
#pragma unroll
  for (int r = 0; r < 8; ++r) {
    float v = mn[r];
#pragma unroll
    for (int off = 8; off; off >>= 1) v = fminf(v, __shfl_xor(v, off, 32));
    mn[r] = v;
  }
  if (lrow == 0) {
#pragma unroll
    for (int r = 0; r < 8; ++r) sMin[wv][8 * half + r] = mn[r];
  }
  __syncthreads();
  if (tid < 16) {
    float v = sMin[0][tid];
#pragma unroll
    for (int w2 = 1; w2 < 8; ++w2) v = fminf(v, sMin[w2][tid]);
    const int m = mt * 16 + tid;
    if (m < M_) dneg[b * M_ + m] = v;
  }
}

// ----------------- kernel 5: final scalar (desc + det losses) ---------------
__global__ void __launch_bounds__(256) finalize_k(const float* __restrict__ dpos,
                                                  const float* __restrict__ dneg,
                                                  const float* __restrict__ sc,
                                                  float* __restrict__ out) {
  __shared__ float lds[8];
  float s1 = 0.f, s2 = 0.f;
  for (int idx = threadIdx.x; idx < B_ * M_; idx += 256) {
    const float dp = dpos[idx], dn = dneg[idx], s = sc[idx];
    s1 += fmaxf(dp - POS_MARGIN, 0.f) + fmaxf(NEG_MARGIN - dn, 0.f);
    s2 += (dp - dn) * s;
  }
  s1 = blockSum(s1, lds);
  s2 = blockSum(s2, lds);
  if (threadIdx.x == 0) out[0] = (s1 + s2) / (float)(B_ * M_);
}

// ---------------------------------------------------------------------------
extern "C" void kernel_launch(void* const* d_in, const int* in_sizes, int n_in,
                              void* d_out, int out_size, void* d_ws, size_t ws_size,
                              hipStream_t stream) {
  (void)in_sizes; (void)n_in; (void)out_size; (void)ws_size;
  const float* x = (const float*)d_in[0];
  const float* y = (const float*)d_in[1];
  const int* stp = (const int*)d_in[2];
  float* out = (float*)d_out;

  // workspace layout (all offsets 256B aligned)
  char* ws = (char*)d_ws;
  size_t off = 0;
  _Float16* y16 = (_Float16*)(ws + off); off += (size_t)B_ * N_ * C_ * 2;     // 16 MB
  _Float16* x16 = (_Float16*)(ws + off); off += (size_t)B_ * MTP_ * C_ * 2;   // 5.4 MB
  float* y2   = (float*)(ws + off); off += (size_t)B_ * N_ * 4;
  float* a2   = (float*)(ws + off); off += (size_t)B_ * MTP_ * 4;
  float* dpos = (float*)(ws + off); off += (size_t)B_ * MTP_ * 4;
  float* dneg = (float*)(ws + off); off += (size_t)B_ * MTP_ * 4;
  float* sc   = (float*)(ws + off); off += (size_t)B_ * MTP_ * 4;

  prep_y<<<dim3(B_ * N_), dim3(256), 0, stream>>>(y, y16, y2);
  prep_x<<<dim3(B_ * MTP_), dim3(256), 0, stream>>>(x, y, stp, x16, a2, dpos);
  score_k<<<dim3(B_ * M_), dim3(256), 0, stream>>>(x, y, stp, sc);
  gemm_dneg<<<dim3(B_ * MT_), dim3(256), 0, stream>>>(x16, y16, a2, y2, stp, dneg);
  finalize_k<<<dim3(1), dim3(256), 0, stream>>>(dpos, dneg, sc, out);
}
